// GCN_51453708206634
// MI455X (gfx1250) — compile-verified, matrix operand order checked by reading, not run on hardware
//
#include <hip/hip_runtime.h>
#include <hip/hip_bf16.h>

typedef __attribute__((ext_vector_type(2)))  float    v2f;
typedef __attribute__((ext_vector_type(8)))  float    v8f;
typedef __attribute__((ext_vector_type(16))) _Float16 v16h;

#define HID 64
#define NCLASS 10

#if defined(__HIP_DEVICE_COMPILE__)
#  if __has_builtin(__builtin_amdgcn_wmma_f32_16x16x4_f32)
#    define GCN_USE_F32_WMMA 1
#  else
#    define GCN_USE_F32_WMMA 0
#  endif
#endif

// Non-returning hardware fp32 atomic add (tracked on STOREcnt; no CAS loop).
__device__ __forceinline__ void atomic_add_f32(float* p, float v) {
    asm volatile("global_atomic_add_f32 %0, %1, off"
                 :
                 : "v"(p), "v"(v)
                 : "memory");
}

// ---------------------------------------------------------------------------
// ew[e] = dot(edge_attr[e, 0:16], We);  deg[row[e]] += ew[e]
// ---------------------------------------------------------------------------
__global__ __launch_bounds__(256) void edge_pre_kernel(
    const float* __restrict__ ea, const float* __restrict__ We,
    const int* __restrict__ rows, float* __restrict__ ew,
    float* __restrict__ deg, int nedge) {
    int e = blockIdx.x * 256 + threadIdx.x;
    if (e >= nedge) return;
    const float4* a4 = (const float4*)(ea + (size_t)e * 16);
    const float4* w4 = (const float4*)We;
    float s = 0.0f;
#pragma unroll
    for (int i = 0; i < 4; ++i) {
        float4 a = a4[i], w = w4[i];
        s += a.x * w.x + a.y * w.y + a.z * w.z + a.w * w.w;
    }
    ew[e] = s;
    atomic_add_f32(&deg[rows[e]], s);
}

__global__ __launch_bounds__(256) void zero_kernel(float* __restrict__ p, int n) {
    int i = blockIdx.x * 256 + threadIdx.x;
    if (i < n) p[i] = 0.0f;
}

// deg -> dinv in place: dinv = rsqrt(deg + 1)
__global__ __launch_bounds__(256) void dinv_kernel(float* __restrict__ deg, int n) {
    int i = blockIdx.x * 256 + threadIdx.x;
    if (i < n) deg[i] = __frsqrt_rn(deg[i] + 1.0f);
}

// ---------------------------------------------------------------------------
// WMMA GEMM: out = A[nrows x K] @ W[K x 64]
//   MODE 0: out0 = h, out1 = dinv[row]^2 * h   (GCN layer + fused self-loop)
//   MODE 1: out0 = h + bias[col]               (linear head)
// Block = 128 threads = 4 waves; wave w owns 16-col tile w; block owns 16 rows.
// ---------------------------------------------------------------------------
template <int K, int MODE>
__global__ __launch_bounds__(128) void gemm_wmma_kernel(
    const float* __restrict__ A, const float* __restrict__ W,
    const float* __restrict__ dinv_or_bias,
    float* __restrict__ out0, float* __restrict__ out1) {
    const int lane = threadIdx.x & 31;
    const int wave = threadIdx.x >> 5;
    const int m    = lane & 15;
    const int hi   = lane >> 4;          // 0 / 1 : lane half
    const int row0 = blockIdx.x * 16;
    const int col0 = wave * 16;

    v8f acc = {0.f, 0.f, 0.f, 0.f, 0.f, 0.f, 0.f, 0.f};

#if defined(__HIP_DEVICE_COMPILE__)
#if GCN_USE_F32_WMMA
    // A 16x4 f32: lane half selects K pair {0,1} / {2,3}  (ISA 7.12.2)
    const int kh = hi * 2;
#pragma unroll 4
    for (int k0 = 0; k0 < K; k0 += 4) {
        const float* ap = A + (size_t)(row0 + m) * K + k0 + kh;
        const float* bp = W + (size_t)(k0 + kh) * HID + col0 + m;
        v2f a, b;
        a.x = ap[0];   a.y = ap[1];      // A[m][k0+kh], A[m][k0+kh+1]
        b.x = bp[0];   b.y = bp[HID];    // W[k0+kh][n], W[k0+kh+1][n]
        acc = __builtin_amdgcn_wmma_f32_16x16x4_f32(
            false, a, false, b, (short)0, acc, false, false);
    }
#else
    // Fallback: convert to f16 in flight, K step 32 (ISA 16-bit A 16x32 layout)
    const int kb = hi * 8;
    for (int k0 = 0; k0 < K; k0 += 32) {
        const float* ap = A + (size_t)(row0 + m) * K + k0 + kb;
        const float* bp = W + (size_t)(k0 + kb) * HID + col0 + m;
        v16h a, b;
#pragma unroll
        for (int j = 0; j < 8; ++j) {
            a[j]     = (_Float16)ap[j];
            a[j + 8] = (_Float16)ap[j + 16];
            b[j]     = (_Float16)bp[(size_t)j * HID];
            b[j + 8] = (_Float16)bp[(size_t)(j + 16) * HID];
        }
        acc = __builtin_amdgcn_wmma_f32_16x16x32_f16(
            false, a, false, b, (short)0, acc, false, false);
    }
#endif
#endif

    // C/D: VGPR r holds M = r + 8*hi, N = lane&15  (ISA 7.12.2)
#pragma unroll
    for (int r = 0; r < 8; ++r) {
        const int row = row0 + r + hi * 8;
        const int col = col0 + m;
        const size_t idx = (size_t)row * HID + col;
        const float v = acc[r];
        if (MODE == 0) {
            out0[idx] = v;
            const float di = dinv_or_bias[row];
            out1[idx] = di * di * v;        // self-loop contribution seeds agg
        } else {
            out0[idx] = v + dinv_or_bias[col];
        }
    }
}

// ---------------------------------------------------------------------------
// Edge scatter: agg[row] += (dinv[row]*ew*dinv[col]) * h[col]
// One half-wave (16 lanes) per edge, float4 per lane (64 floats = 256B).
// Gather is one coalesced b128 load per lane; scatter is 4 hardware
// global_atomic_add_f32 per lane into L2-resident agg.
// ---------------------------------------------------------------------------
__global__ __launch_bounds__(256) void edge_agg_kernel(
    const int* __restrict__ rows, const int* __restrict__ cols,
    const float* __restrict__ ew, const float* __restrict__ dinv,
    const float* __restrict__ h, float* __restrict__ agg, int nedge) {
    const int e = blockIdx.x * 16 + (threadIdx.x >> 4);   // 16 edges / block
    if (e >= nedge) return;
    const int l = threadIdx.x & 15;
    const int r = rows[e];
    const int c = cols[e];
    const float nrm = dinv[r] * ew[e] * dinv[c];
    const float4 hv = *(const float4*)(h + (size_t)c * HID + l * 4);
    float* dst = agg + (size_t)r * HID + l * 4;
    atomic_add_f32(dst + 0, nrm * hv.x);
    atomic_add_f32(dst + 1, nrm * hv.y);
    atomic_add_f32(dst + 2, nrm * hv.z);
    atomic_add_f32(dst + 3, nrm * hv.w);
}

// x_next = relu(agg + b)
__global__ __launch_bounds__(256) void finalize_kernel(
    const float* __restrict__ agg, const float* __restrict__ bias,
    float* __restrict__ xout, int total) {
    int i = blockIdx.x * 256 + threadIdx.x;
    if (i >= total) return;
    float v = agg[i] + bias[i & (HID - 1)];
    xout[i] = v > 0.0f ? v : 0.0f;
}

// out[n, 0:10] = t[n, 0:64] @ lin2_W + lin2_b   (tiny; VALU)
__global__ __launch_bounds__(256) void head2_kernel(
    const float* __restrict__ t, const float* __restrict__ W,
    const float* __restrict__ b, float* __restrict__ out, int n) {
    int i = blockIdx.x * 256 + threadIdx.x;
    if (i >= n) return;
    float acc[NCLASS];
#pragma unroll
    for (int c = 0; c < NCLASS; ++c) acc[c] = b[c];
    const float* ti = t + (size_t)i * HID;
#pragma unroll 4
    for (int j = 0; j < HID; ++j) {
        const float tv = ti[j];
        const float* wr = W + (size_t)j * NCLASS;
#pragma unroll
        for (int c = 0; c < NCLASS; ++c) acc[c] += tv * wr[c];
    }
#pragma unroll
    for (int c = 0; c < NCLASS; ++c) out[(size_t)i * NCLASS + c] = acc[c];
}

// ---------------------------------------------------------------------------
extern "C" void kernel_launch(void* const* d_in, const int* in_sizes, int n_in,
                              void* d_out, int out_size, void* d_ws, size_t ws_size,
                              hipStream_t stream) {
    const float* x      = (const float*)d_in[0];    // [N,128]
    const float* eattr  = (const float*)d_in[1];    // [E,16]
    const int*   eidx   = (const int*)  d_in[2];    // [2,E]
    const float* We     = (const float*)d_in[3];    // [16,1]
    const float* Wl[3]  = {(const float*)d_in[4], (const float*)d_in[6], (const float*)d_in[8]};
    const float* bl[3]  = {(const float*)d_in[5], (const float*)d_in[7], (const float*)d_in[9]};
    const float* lin1_W = (const float*)d_in[10];
    const float* lin1_b = (const float*)d_in[11];
    const float* lin2_W = (const float*)d_in[12];
    const float* lin2_b = (const float*)d_in[13];
    float* out = (float*)d_out;

    const int N = in_sizes[0] / 128;
    const int E = in_sizes[1] / 16;
    const int* rows = eidx;
    const int* cols = eidx + E;

    float* ws   = (float*)d_ws;
    float* ew   = ws;                         // E
    float* dinv = ew + E;                     // N (deg -> dinv in place)
    float* h    = dinv + N;                   // N*64
    float* agg  = h + (size_t)N * HID;        // N*64
    float* xa   = agg + (size_t)N * HID;      // N*64
    float* xb   = xa + (size_t)N * HID;       // N*64
    float* t    = h;                          // head1 output reuses h

    const int rowTiles = (N + 15) / 16;       // N = 50000 -> 3125 (exact)
    const int eBlk     = (E + 15) / 16;       // 16 edges per 256-thread block
    const int nodeElts = N * HID;

    // edge weights + symmetric-norm degrees
    zero_kernel<<<(N + 255) / 256, 256, 0, stream>>>(dinv, N);
    edge_pre_kernel<<<(E + 255) / 256, 256, 0, stream>>>(eattr, We, rows, ew, dinv, E);
    dinv_kernel<<<(N + 255) / 256, 256, 0, stream>>>(dinv, N);

    // layer 0: K = 128
    gemm_wmma_kernel<128, 0><<<rowTiles, 128, 0, stream>>>(x, Wl[0], dinv, h, agg);
    edge_agg_kernel<<<eBlk, 256, 0, stream>>>(rows, cols, ew, dinv, h, agg, E);
    finalize_kernel<<<(nodeElts + 255) / 256, 256, 0, stream>>>(agg, bl[0], xa, nodeElts);

    // layer 1: K = 64
    gemm_wmma_kernel<64, 0><<<rowTiles, 128, 0, stream>>>(xa, Wl[1], dinv, h, agg);
    edge_agg_kernel<<<eBlk, 256, 0, stream>>>(rows, cols, ew, dinv, h, agg, E);
    finalize_kernel<<<(nodeElts + 255) / 256, 256, 0, stream>>>(agg, bl[1], xb, nodeElts);

    // layer 2: K = 64
    gemm_wmma_kernel<64, 0><<<rowTiles, 128, 0, stream>>>(xb, Wl[2], dinv, h, agg);
    edge_agg_kernel<<<eBlk, 256, 0, stream>>>(rows, cols, ew, dinv, h, agg, E);
    finalize_kernel<<<(nodeElts + 255) / 256, 256, 0, stream>>>(agg, bl[2], xa, nodeElts);

    // head: lin1 (WMMA, bias fused), then lin2 (tiny VALU)
    gemm_wmma_kernel<64, 1><<<rowTiles, 128, 0, stream>>>(xa, lin1_W, lin1_b, t, nullptr);
    head2_kernel<<<(N + 255) / 256, 256, 0, stream>>>(t, lin2_W, lin2_b, out, N);
}
